// SharedEmbeddingDQNGNN_20907900797453
// MI455X (gfx1250) — compile-verified, matrix operand order checked by reading, not run on hardware
//
#include <hip/hip_runtime.h>
#include <hip/hip_bf16.h>

typedef __attribute__((ext_vector_type(16))) _Float16 v16h;
typedef __attribute__((ext_vector_type(8)))  _Float16 v8h;
typedef __attribute__((ext_vector_type(8)))  float    v8f;

// ---------------------------------------------------------------- utilities
__global__ void fill_f32(float* __restrict__ p, float v, long long n) {
    long long i = (long long)blockIdx.x * blockDim.x + threadIdx.x;
    if (i < n) p[i] = v;
}

__global__ void deg_accum(const int* __restrict__ dst, float* __restrict__ deg, int E) {
    int e = blockIdx.x * blockDim.x + threadIdx.x;
    if (e < E) atomicAdd(&deg[dst[e]], 1.0f);
}

__global__ void rsqrt_inplace(float* __restrict__ d, int n) {
    int i = blockIdx.x * blockDim.x + threadIdx.x;
    if (i < n) {
        float v = d[i];
        d[i] = (v > 0.0f) ? rsqrtf(v) : 0.0f;
    }
}

__global__ void cvt_f32_f16(const float* __restrict__ src, _Float16* __restrict__ dst, long long n) {
    long long i = (long long)blockIdx.x * blockDim.x + threadIdx.x;
    if (i < n) dst[i] = (_Float16)src[i];
}

// W: K x C row-major (f32)  ->  Wt: C x K row-major (f16)   (i.e. B stored column-major)
__global__ void cvt_transpose_w(const float* __restrict__ W, _Float16* __restrict__ Wt, int K, int C) {
    int idx = blockIdx.x * blockDim.x + threadIdx.x;
    if (idx < K * C) {
        int k = idx / C, c = idx % C;
        Wt[c * K + k] = (_Float16)W[idx];
    }
}

// --------------------------------------------------------------- WMMA GEMM
// C[M,128] = A[M,128] @ B[128,128], A f16 row-major, Bt = B^T f16 row-major.
// Block = 256 threads = 8 waves. Block -> 16 rows (m-tile), wave w -> column tile w.
// Per-lane fragment addressing follows the CDNA5 16-bit A/B layout:
//   lanes 0-15 : K = {kc*32 + 0..7, kc*32 + 16..23}
//   lanes 16-31: K = {kc*32 + 8..15, kc*32 + 24..31}
// -> two contiguous 16-byte loads per fragment per k-chunk.
__global__ void __launch_bounds__(256) gemm_f16_wmma(const _Float16* __restrict__ A,
                                                     const _Float16* __restrict__ Bt,
                                                     float* __restrict__ C,
                                                     int nrows) {
    const int wave = threadIdx.x >> 5;       // 0..7 -> n tile
    const int lane = threadIdx.x & 31;
    const int mt   = blockIdx.x;

    int rowA = mt * 16 + (lane & 15);
    if (rowA > nrows - 1) rowA = nrows - 1;  // clamp loads; WMMA keeps EXEC all-ones
    const int colB  = wave * 16 + (lane & 15);
    const int kbase = (lane >> 4) ? 8 : 0;

    const _Float16* __restrict__ arow = A  + (long long)rowA * 128;
    const _Float16* __restrict__ brow = Bt + (long long)colB * 128;

    v8f acc = {};
#pragma unroll
    for (int kc = 0; kc < 4; ++kc) {
        const int k0 = kc * 32 + kbase;
        v8h alo = *(const v8h*)(arow + k0);
        v8h ahi = *(const v8h*)(arow + k0 + 16);
        v8h blo = *(const v8h*)(brow + k0);
        v8h bhi = *(const v8h*)(brow + k0 + 16);
        v16h a, b;
#pragma unroll
        for (int i = 0; i < 8; ++i) {
            a[i] = alo[i]; a[8 + i] = ahi[i];
            b[i] = blo[i]; b[8 + i] = bhi[i];
        }
        acc = __builtin_amdgcn_wmma_f32_16x16x32_f16(
            /*neg_a=*/false, a, /*neg_b=*/false, b,
            /*c_mod=*/(short)0, acc, /*reuse_a=*/false, /*reuse_b=*/false);
    }

    // C layout: VGPR r -> M = r (lanes 0-15) / M = 8+r (lanes 16-31), N = lane&15.
    const int rbase = mt * 16 + ((lane >> 4) ? 8 : 0);
    float* __restrict__ crow = C + (long long)rbase * 128 + wave * 16 + (lane & 15);
    if (mt * 16 + 16 <= nrows) {
        // full tile (uniform branch): 8 unguarded stores, clause-friendly
#pragma unroll
        for (int r = 0; r < 8; ++r) crow[(long long)r * 128] = acc[r];
    } else {
        // ragged last tile only
#pragma unroll
        for (int r = 0; r < 8; ++r)
            if (rbase + r < nrows) crow[(long long)r * 128] = acc[r];
    }
}

// ------------------------------------------------------- sparse aggregation
// One thread handles (edge, 4-feature chunk). Edges e >= E are the self-loops.
__global__ void scatter_edges(const int* __restrict__ ei, const float* __restrict__ dis,
                              const float* __restrict__ hin, float* __restrict__ agg,
                              int E, int Nn) {
    long long idx = (long long)blockIdx.x * blockDim.x + threadIdx.x;
    long long total = (long long)(E + Nn) * 32;
    if (idx >= total) return;
    int e = (int)(idx >> 5);
    int c = ((int)idx & 31) * 4;
    int s, d;
    if (e < E) { s = ei[e]; d = ei[E + e]; }
    else       { s = d = e - E; }
    float w = dis[s] * dis[d];
    float4 v = *(const float4*)(hin + (long long)s * 128 + c);
    float* o = agg + (long long)d * 128 + c;
    atomicAdd(o + 0, v.x * w);
    atomicAdd(o + 1, v.y * w);
    atomicAdd(o + 2, v.z * w);
    atomicAdd(o + 3, v.w * w);
}

__global__ void bias_relu_f16(const float* __restrict__ agg, const float* __restrict__ b,
                              _Float16* __restrict__ out, long long n) {
    long long i = (long long)blockIdx.x * blockDim.x + threadIdx.x;
    if (i < n) {
        float v = agg[i] + b[i & 127];
        out[i] = (_Float16)(v > 0.0f ? v : 0.0f);
    }
}

__global__ void bias_relu_f32(const float* __restrict__ agg, const float* __restrict__ b,
                              float* __restrict__ out, long long n) {
    long long i = (long long)blockIdx.x * blockDim.x + threadIdx.x;
    if (i < n) {
        float v = agg[i] + b[i & 127];
        out[i] = v > 0.0f ? v : 0.0f;
    }
}

// -------------------------------------------------------------------- Q head
__global__ void q_kernel(const int* __restrict__ va, const float* __restrict__ h,
                         const float* __restrict__ Wq, const float* __restrict__ bq,
                         float* __restrict__ out, int A) {
    int a = blockIdx.x * blockDim.x + threadIdx.x;
    if (a >= A) return;
    int i = va[2 * a], j = va[2 * a + 1];
    const float4* __restrict__ hi = (const float4*)(h + (long long)i * 128);
    const float4* __restrict__ hj = (const float4*)(h + (long long)j * 128);
    const float4* __restrict__ wq = (const float4*)Wq;
    float s = bq[0];
#pragma unroll 8
    for (int k = 0; k < 32; ++k) {
        float4 x = hi[k], w = wq[k];
        s += x.x * w.x + x.y * w.y + x.z * w.z + x.w * w.w;
    }
#pragma unroll 8
    for (int k = 0; k < 32; ++k) {
        float4 x = hj[k], w = wq[32 + k];
        s += x.x * w.x + x.y * w.y + x.z * w.z + x.w * w.w;
    }
    out[a] = s;
}

// ---------------------------------------------------------------------------
extern "C" void kernel_launch(void* const* d_in, const int* in_sizes, int n_in,
                              void* d_out, int out_size, void* d_ws, size_t ws_size,
                              hipStream_t stream) {
    const float* x  = (const float*)d_in[0];
    const int*   ei = (const int*)d_in[1];
    const int*   va = (const int*)d_in[2];
    const float* W1 = (const float*)d_in[3];
    const float* b1 = (const float*)d_in[4];
    const float* W2 = (const float*)d_in[5];
    const float* b2 = (const float*)d_in[6];
    const float* Wq = (const float*)d_in[7];
    const float* bq = (const float*)d_in[8];
    float* q = (float*)d_out;

    const int H  = 128;
    const int Nn = in_sizes[0] / H;   // 50000
    const int E  = in_sizes[1] / 2;   // 600000
    const int A  = in_sizes[2] / 2;   // 200000

    // ---- carve workspace (all 256B aligned) ----
    char*  ws  = (char*)d_ws;
    size_t off = 0;
    auto alloc = [&](size_t bytes) -> char* {
        char* p = ws + off;
        off = (off + bytes + 255) & ~(size_t)255;
        return p;
    };
    _Float16* xh  = (_Float16*)alloc((size_t)Nn * H * 2);   // x in f16
    _Float16* w1t = (_Float16*)alloc((size_t)H * H * 2);    // W1^T f16
    _Float16* w2t = (_Float16*)alloc((size_t)H * H * 2);    // W2^T f16
    float*    gout= (float*)   alloc((size_t)Nn * H * 4);   // GEMM out / reused as h2
    float*    agg = (float*)   alloc((size_t)Nn * H * 4);   // scatter target
    _Float16* h1h = (_Float16*)alloc((size_t)Nn * H * 2);   // relu(layer1) f16
    float*    dis = (float*)   alloc((size_t)Nn * 4);       // deg -> rsqrt(deg)
    float*    h2  = gout;                                   // reuse after scatter2

    const long long NH = (long long)Nn * H;
    auto blks = [](long long n, int t) { return (unsigned)((n + t - 1) / t); };

    // degree / norm
    fill_f32<<<blks(Nn, 256), 256, 0, stream>>>(dis, 1.0f, Nn);          // self-loop
    deg_accum<<<blks(E, 256), 256, 0, stream>>>(ei + E, dis, E);         // dst row
    rsqrt_inplace<<<blks(Nn, 256), 256, 0, stream>>>(dis, Nn);

    // f16 conversions (W stored transposed so WMMA B-fragments are contiguous)
    cvt_f32_f16<<<blks(NH, 256), 256, 0, stream>>>(x, xh, NH);
    cvt_transpose_w<<<blks(H * H, 256), 256, 0, stream>>>(W1, w1t, H, H);
    cvt_transpose_w<<<blks(H * H, 256), 256, 0, stream>>>(W2, w2t, H, H);

    const unsigned mtiles = (unsigned)((Nn + 15) / 16);
    const long long scat  = (long long)(E + Nn) * 32;

    // layer 1
    gemm_f16_wmma<<<mtiles, 256, 0, stream>>>(xh, w1t, gout, Nn);
    fill_f32<<<blks(NH, 256), 256, 0, stream>>>(agg, 0.0f, NH);
    scatter_edges<<<blks(scat, 256), 256, 0, stream>>>(ei, dis, gout, agg, E, Nn);
    bias_relu_f16<<<blks(NH, 256), 256, 0, stream>>>(agg, b1, h1h, NH);

    // layer 2
    gemm_f16_wmma<<<mtiles, 256, 0, stream>>>(h1h, w2t, gout, Nn);
    fill_f32<<<blks(NH, 256), 256, 0, stream>>>(agg, 0.0f, NH);
    scatter_edges<<<blks(scat, 256), 256, 0, stream>>>(ei, dis, gout, agg, E, Nn);
    bias_relu_f32<<<blks(NH, 256), 256, 0, stream>>>(agg, b2, h2, NH);

    // Q head
    q_kernel<<<blks(A, 256), 256, 0, stream>>>(va, h2, Wq, bq, q, A);
}
